// TransitionModel_64871186038795
// MI455X (gfx1250) — matrix operand order repaired and checked by reading, not compiled, not run on hardware
//
#include <hip/hip_runtime.h>
#include <hip/hip_bf16.h>
#include <math.h>

// ---------------- problem dims (fixed by the reference) ----------------
#define BSZ    512
#define TM1    64
#define BELIEF 2048
#define STATE  256
#define ACTION 64
#define HID    2048
#define EMB    1024
#define MIN_STD 0.1f

typedef unsigned short ushort_t;

// ---------------- WMMA types ----------------
typedef __attribute__((ext_vector_type(16))) __bf16 v16bf;
typedef __attribute__((ext_vector_type(8)))  float  v8f;
typedef __attribute__((ext_vector_type(4)))  unsigned v4u;

union FragU { v16bf v; v4u q[2]; };

__device__ __forceinline__ ushort_t f2bf(float f) {
  unsigned x = __float_as_uint(f);
  unsigned r = (x + 0x7FFFu + ((x >> 16) & 1u)) >> 16;   // RNE
  return (ushort_t)r;
}

// ---------------- GEMM ----------------
// 256 threads = 8 wave32, arranged 2(M) x 4(N). BK=32.
// Wave tile: 64 x (16*FN).  Block tile: 128 x (64*FN).
//   FN=4 -> 16 wmma / 16 b128 loads per K-step (high arithmetic intensity)
//   FN=2 -> smaller tiles for narrow-N GEMMs (more blocks in flight)
// Operands are bf16 in global memory in fragment-native layouts:
//   A  : [M][lda] row-major bf16 -> frag = two b128 loads (k0+h*8, k0+16+h*8)
//   Bt : [N][K]   row-major bf16 -> frag = two b128 loads (k0+h*8, k0+16+h*8)
// No LDS, no barriers: all operands are L2-resident.
#define BM 128
#define BK 32

template <int FN>
__global__ __launch_bounds__(256) void gemm_bf16_direct(
    const ushort_t* __restrict__ A, int lda,        // bf16 [M][lda]
    const ushort_t* __restrict__ Bt,                // bf16 [N][K]
    const float* __restrict__ bias,                 // [N]
    float*    __restrict__ Cf,                      // fp32 out (nullable)
    ushort_t* __restrict__ Cb, int ldcb,            // bf16 out (nullable)
    int N, int K, int relu)
{
  const int tid  = threadIdx.x;
  const int lane = tid & 31;
  const int wave = tid >> 5;
  const int wm   = wave >> 2;      // 0..1
  const int wn   = wave & 3;       // 0..3
  const int hl   = lane >> 4;      // lane half
  const int l16  = lane & 15;
  const int m0   = blockIdx.y * BM;
  const int n0   = blockIdx.x * (64 * FN);

  const ushort_t* aptr[4];
#pragma unroll
  for (int fm = 0; fm < 4; ++fm)
    aptr[fm] = A + (size_t)(m0 + wm * 64 + fm * 16 + l16) * lda + hl * 8;
  const ushort_t* bptr[FN];
#pragma unroll
  for (int fn = 0; fn < FN; ++fn)
    bptr[fn] = Bt + (size_t)(n0 + (wn * FN + fn) * 16 + l16) * K + hl * 8;

  v8f acc[4][FN];
#pragma unroll
  for (int i = 0; i < 4; ++i)
#pragma unroll
    for (int j = 0; j < FN; ++j)
      acc[i][j] = (v8f){0.f,0.f,0.f,0.f,0.f,0.f,0.f,0.f};

  for (int k0 = 0; k0 < K; k0 += BK) {
    FragU af[4], bf[FN];
#pragma unroll
    for (int fm = 0; fm < 4; ++fm) {
      af[fm].q[0] = *(const v4u*)(aptr[fm] + k0);        // k = hl*8 + 2v      (v=0..3)
      af[fm].q[1] = *(const v4u*)(aptr[fm] + k0 + 16);   // k = 16+hl*8+2(v-4) (v=4..7)
    }
#pragma unroll
    for (int fn = 0; fn < FN; ++fn) {
      bf[fn].q[0] = *(const v4u*)(bptr[fn] + k0);        // k = hl*8 + 2v
      bf[fn].q[1] = *(const v4u*)(bptr[fn] + k0 + 16);
    }
#pragma unroll
    for (int fm = 0; fm < 4; ++fm)
#pragma unroll
      for (int fn = 0; fn < FN; ++fn)
        acc[fm][fn] = __builtin_amdgcn_wmma_f32_16x16x32_bf16(
            false, af[fm].v, false, bf[fn].v, (short)0, acc[fm][fn], false, false);
  }

  // ---- epilogue: bias (+ReLU); store fp32 and/or bf16 ----
#pragma unroll
  for (int fm = 0; fm < 4; ++fm)
#pragma unroll
    for (int fn = 0; fn < FN; ++fn) {
      int n = n0 + (wn * FN + fn) * 16 + l16;
      float bv = bias ? bias[n] : 0.f;
#pragma unroll
      for (int v = 0; v < 8; ++v) {
        int m = m0 + wm * 64 + fm * 16 + hl * 8 + v;     // C/D f32 layout
        float r = acc[fm][fn][v] + bv;
        if (relu) r = fmaxf(r, 0.f);
        if (Cf) Cf[(size_t)m * N + n] = r;
        if (Cb) Cb[(size_t)m * ldcb + n] = f2bf(r);
      }
    }
}

// ---------------- weight convert + transpose: Wt[n][k] = bf16(W[k][n]) ----------------
__global__ __launch_bounds__(256) void conv_transpose_bf16(
    const float* __restrict__ W, ushort_t* __restrict__ Wt, int K, int N)
{
  __shared__ ushort_t tile[32][33];
  const int kb = blockIdx.y * 32, nb = blockIdx.x * 32;
  const int tx = threadIdx.x & 31, ty = threadIdx.x >> 5;   // 32x8
#pragma unroll
  for (int i = ty; i < 32; i += 8)
    tile[i][tx] = f2bf(W[(size_t)(kb + i) * N + nb + tx]);
  __syncthreads();
#pragma unroll
  for (int i = ty; i < 32; i += 8)
    Wt[(size_t)(nb + i) * K + kb + tx] = tile[tx][i];
}

// ---------------- per-step activation prep (fp32 -> bf16 in consumer layout) ----------
__global__ void prep_embed(const float* __restrict__ s, const float* __restrict__ a,
                           const float* __restrict__ nt, ushort_t* __restrict__ Aemb)
{
  int i = blockIdx.x * blockDim.x + threadIdx.x;
  if (i >= BSZ * (STATE + ACTION)) return;
  int b = i / (STATE + ACTION), c = i - b * (STATE + ACTION);
  float v = (c < STATE) ? s[(size_t)b * STATE + c] * nt[b]
                        : a[(size_t)b * ACTION + (c - STATE)];
  Aemb[i] = f2bf(v);
}

__global__ void prep_obs(const float* __restrict__ obs, ushort_t* __restrict__ Aq)
{
  int i = blockIdx.x * blockDim.x + threadIdx.x;
  if (i >= BSZ * EMB) return;
  int b = i / EMB, c = i - b * EMB;
  Aq[(size_t)b * (BELIEF + EMB) + BELIEF + c] = f2bf(obs[i]);
}

__global__ void prep_belief0(const float* __restrict__ bel, ushort_t* __restrict__ Aq)
{
  int i = blockIdx.x * blockDim.x + threadIdx.x;
  if (i >= BSZ * BELIEF) return;
  int b = i / BELIEF, c = i - b * BELIEF;
  Aq[(size_t)b * (BELIEF + EMB) + c] = f2bf(bel[i]);
}

// ---------------- GRU gate combine -> new belief (fp32 out + bf16 into Aq) -----------
__global__ void gru_combine(const float* __restrict__ gi, const float* __restrict__ gh,
                            const float* __restrict__ bprev, float* __restrict__ bnew,
                            ushort_t* __restrict__ Aq)
{
  int i = blockIdx.x * blockDim.x + threadIdx.x;
  if (i >= BSZ * BELIEF) return;
  int b = i / BELIEF, h = i - b * BELIEF;
  const size_t H3 = (size_t)3 * BELIEF;
  float ir = gi[b * H3 + h];
  float iz = gi[b * H3 + BELIEF + h];
  float in_ = gi[b * H3 + 2 * BELIEF + h];
  float hr = gh[b * H3 + h];
  float hz = gh[b * H3 + BELIEF + h];
  float hn = gh[b * H3 + 2 * BELIEF + h];
  float r = 1.f / (1.f + __expf(-(ir + hr)));
  float z = 1.f / (1.f + __expf(-(iz + hz)));
  float n = tanhf(in_ + r * hn);
  float nb = (1.f - z) * n + z * bprev[i];
  bnew[i] = nb;
  Aq[(size_t)b * (BELIEF + EMB) + h] = f2bf(nb);
}

// ---------------- rsample: loc + (softplus(raw)+0.1) * eps ----------------
__global__ void sample_k(const float* __restrict__ dist, const float* __restrict__ eps,
                         float* __restrict__ states, float* __restrict__ means,
                         float* __restrict__ stds)
{
  int i = blockIdx.x * blockDim.x + threadIdx.x;
  if (i >= BSZ * STATE) return;
  int b = i / STATE, s = i - b * STATE;
  float loc = dist[(size_t)b * 2 * STATE + s];
  float raw = dist[(size_t)b * 2 * STATE + STATE + s];
  float sp = fmaxf(raw, 0.f) + log1pf(__expf(-fabsf(raw)));   // stable softplus
  float sc = sp + MIN_STD;
  means[i]  = loc;
  stds[i]   = sc;
  states[i] = loc + sc * eps[i];
}

// ---------------- driver ----------------
extern "C" void kernel_launch(void* const* d_in, const int* in_sizes, int n_in,
                              void* d_out, int out_size, void* d_ws, size_t ws_size,
                              hipStream_t stream) {
  (void)in_sizes; (void)n_in; (void)out_size; (void)ws_size;

  const float* prev_state  = (const float*)d_in[0];
  const float* actions     = (const float*)d_in[1];
  const float* prev_belief = (const float*)d_in[2];
  const float* obs_emb     = (const float*)d_in[3];
  const float* nonterm     = (const float*)d_in[4];
  const float* W_embed     = (const float*)d_in[5];
  const float* b_embed     = (const float*)d_in[6];
  const float* Wi          = (const float*)d_in[7];
  const float* bi          = (const float*)d_in[8];
  const float* Wh          = (const float*)d_in[9];
  const float* bh          = (const float*)d_in[10];
  const float* W1          = (const float*)d_in[11];
  const float* b1          = (const float*)d_in[12];
  const float* W2          = (const float*)d_in[13];
  const float* b2          = (const float*)d_in[14];
  const float* W1p         = (const float*)d_in[15];
  const float* b1p         = (const float*)d_in[16];
  const float* W2p         = (const float*)d_in[17];
  const float* b2p         = (const float*)d_in[18];
  const float* eps_prior   = (const float*)d_in[19];
  const float* eps_post    = (const float*)d_in[20];

  float* out = (float*)d_out;
  const size_t nBel = (size_t)TM1 * BSZ * BELIEF;
  const size_t nSt  = (size_t)TM1 * BSZ * STATE;
  float* beliefs      = out;
  float* prior_states = beliefs + nBel;
  float* prior_means  = prior_states + nSt;
  float* prior_stds   = prior_means + nSt;
  float* post_states  = prior_stds + nSt;
  float* post_means   = post_states + nSt;
  float* post_stds    = post_means + nSt;

  // ---- workspace layout: bf16 region, then fp32 scratch ----
  ushort_t* wsb = (ushort_t*)d_ws;
  ushort_t* Wt_embed = wsb;                                    // [2048][320]
  ushort_t* Wt_i     = Wt_embed + (size_t)BELIEF * 320;        // [6144][2048]
  ushort_t* Wt_h     = Wt_i     + (size_t)3 * BELIEF * BELIEF; // [6144][2048]
  ushort_t* Wt_1     = Wt_h     + (size_t)3 * BELIEF * BELIEF; // [2048][2048]
  ushort_t* Wt_2     = Wt_1     + (size_t)HID * BELIEF;        // [512][2048]
  ushort_t* Wt_1p    = Wt_2     + (size_t)2 * STATE * HID;     // [2048][3072]
  ushort_t* Wt_2p    = Wt_1p    + (size_t)HID * (BELIEF+EMB);  // [512][2048]
  ushort_t* Aemb     = Wt_2p    + (size_t)2 * STATE * HID;     // [512][320]
  ushort_t* hid_bf   = Aemb     + (size_t)BSZ * (STATE+ACTION);// [512][2048]
  ushort_t* Aq       = hid_bf   + (size_t)BSZ * BELIEF;        // [512][3072]
  ushort_t* hp_bf    = Aq       + (size_t)BSZ * (BELIEF+EMB);  // [512][2048]
  ushort_t* hq_bf    = hp_bf    + (size_t)BSZ * HID;           // [512][2048]
  float*    wsf      = (float*)(hq_bf + (size_t)BSZ * HID);
  float*    gi       = wsf;                                    // [512][6144]
  float*    gh       = gi + (size_t)BSZ * 3 * BELIEF;          // [512][6144]
  float*    dp       = gh + (size_t)BSZ * 3 * BELIEF;          // [512][512]
  float*    dq       = dp + (size_t)BSZ * 2 * STATE;           // [512][512]

  const dim3 blk(256);
  const dim3 grdEw((BSZ * BELIEF + 255) / 256);
  const dim3 grdSm((BSZ * STATE + 255) / 256);

  // ---- per-launch weight convert+transpose into bf16 [N][K] ----
  conv_transpose_bf16<<<dim3(BELIEF/32,   320/32), blk, 0, stream>>>(W_embed, Wt_embed, 320, BELIEF);
  conv_transpose_bf16<<<dim3(3*BELIEF/32, BELIEF/32), blk, 0, stream>>>(Wi, Wt_i, BELIEF, 3*BELIEF);
  conv_transpose_bf16<<<dim3(3*BELIEF/32, BELIEF/32), blk, 0, stream>>>(Wh, Wt_h, BELIEF, 3*BELIEF);
  conv_transpose_bf16<<<dim3(HID/32,      BELIEF/32), blk, 0, stream>>>(W1, Wt_1, BELIEF, HID);
  conv_transpose_bf16<<<dim3(2*STATE/32,  HID/32), blk, 0, stream>>>(W2, Wt_2, HID, 2*STATE);
  conv_transpose_bf16<<<dim3(HID/32, (BELIEF+EMB)/32), blk, 0, stream>>>(W1p, Wt_1p, BELIEF+EMB, HID);
  conv_transpose_bf16<<<dim3(2*STATE/32,  HID/32), blk, 0, stream>>>(W2p, Wt_2p, HID, 2*STATE);
  prep_belief0<<<grdEw, blk, 0, stream>>>(prev_belief, Aq);

  for (int t = 0; t < TM1; ++t) {
    const float* sPrev = (t == 0) ? prev_state  : post_states + (size_t)(t - 1) * BSZ * STATE;
    const float* bPrev = (t == 0) ? prev_belief : beliefs     + (size_t)(t - 1) * BSZ * BELIEF;
    const float* at = actions + (size_t)t * BSZ * ACTION;
    const float* ot = obs_emb + (size_t)t * BSZ * EMB;
    const float* nt = nonterm + (size_t)t * BSZ;
    float* bNew = beliefs + (size_t)t * BSZ * BELIEF;

    // 0) build [s*nt, a] bf16
    prep_embed<<<(BSZ*(STATE+ACTION)+255)/256, blk, 0, stream>>>(sPrev, at, nt, Aemb);
    // 1) hidden(bf16) = relu(Aemb @ W_embed + b_embed)     K=320, N=2048
    gemm_bf16_direct<4><<<dim3(BELIEF/256, BSZ/BM), blk, 0, stream>>>(
        Aemb, STATE+ACTION, Wt_embed, b_embed, nullptr, hid_bf, BELIEF,
        BELIEF, STATE+ACTION, 1);
    // 2) gi(fp32) = hidden @ Wi + bi                       K=2048, N=6144
    gemm_bf16_direct<4><<<dim3(3*BELIEF/256, BSZ/BM), blk, 0, stream>>>(
        hid_bf, BELIEF, Wt_i, bi, gi, nullptr, 0, 3*BELIEF, BELIEF, 0);
    // 3) gh(fp32) = belief(bf16, in Aq from prev step) @ Wh + bh
    gemm_bf16_direct<4><<<dim3(3*BELIEF/256, BSZ/BM), blk, 0, stream>>>(
        Aq, BELIEF+EMB, Wt_h, bh, gh, nullptr, 0, 3*BELIEF, BELIEF, 0);
    // 4) GRU gates -> new belief: fp32 into d_out, bf16 into Aq[:, :2048]
    gru_combine<<<grdEw, blk, 0, stream>>>(gi, gh, bPrev, bNew, Aq);
    // 4b) obs bf16 into Aq[:, 2048:3072]
    prep_obs<<<(BSZ*EMB+255)/256, blk, 0, stream>>>(ot, Aq);
    // 5) hp(bf16) = relu(new_belief @ W1 + b1)
    gemm_bf16_direct<4><<<dim3(HID/256, BSZ/BM), blk, 0, stream>>>(
        Aq, BELIEF+EMB, Wt_1, b1, nullptr, hp_bf, HID, HID, BELIEF, 1);
    // 6) dp(fp32) = hp @ W2 + b2                           N=512 (narrow -> FN=2)
    gemm_bf16_direct<2><<<dim3(2*STATE/128, BSZ/BM), blk, 0, stream>>>(
        hp_bf, HID, Wt_2, b2, dp, nullptr, 0, 2*STATE, HID, 0);
    // 7) prior rsample
    sample_k<<<grdSm, blk, 0, stream>>>(dp, eps_prior + (size_t)t * BSZ * STATE,
        prior_states + (size_t)t * BSZ * STATE,
        prior_means  + (size_t)t * BSZ * STATE,
        prior_stds   + (size_t)t * BSZ * STATE);
    // 8) hq(bf16) = relu([new_belief, obs] @ W1p + b1p)    K=3072
    gemm_bf16_direct<4><<<dim3(HID/256, BSZ/BM), blk, 0, stream>>>(
        Aq, BELIEF+EMB, Wt_1p, b1p, nullptr, hq_bf, HID, HID, BELIEF+EMB, 1);
    // 9) dq(fp32) = hq @ W2p + b2p
    gemm_bf16_direct<2><<<dim3(2*STATE/128, BSZ/BM), blk, 0, stream>>>(
        hq_bf, HID, Wt_2p, b2p, dq, nullptr, 0, 2*STATE, HID, 0);
    // 10) posterior rsample (feeds next step)
    sample_k<<<grdSm, blk, 0, stream>>>(dq, eps_post + (size_t)t * BSZ * STATE,
        post_states + (size_t)t * BSZ * STATE,
        post_means  + (size_t)t * BSZ * STATE,
        post_stds   + (size_t)t * BSZ * STATE);
  }
}
